// SABlock_28647431864330
// MI455X (gfx1250) — matrix operand order repaired and checked by reading, not compiled, not run on hardware
//
#include <hip/hip_runtime.h>
#include <hip/hip_bf16.h>

typedef __attribute__((ext_vector_type(16))) __bf16 v16bf;
typedef __attribute__((ext_vector_type(8))) float v8f;
typedef __attribute__((ext_vector_type(4))) int i32x4;

struct bf16x4 { __bf16 a, b, c, d; };
struct bf16x8 { __bf16 v[8]; };
struct u4x2 { uint4 lo, hi; };

#define HDIM 768
#define SEQ 2048
#define NHEADS 12
#define DHEAD 64

#if defined(__has_builtin)
#if __has_builtin(__builtin_amdgcn_global_load_async_to_lds_b128)
#define SAB_ASYNC 1
#endif
#endif

#ifdef SAB_ASYNC
typedef __attribute__((address_space(1))) i32x4* as1_v4i;
typedef __attribute__((address_space(3))) i32x4* as3_v4i;
__device__ __forceinline__ void gasync_b128(const __bf16* g, __bf16* l) {
  i32x4* gg = (i32x4*)(void*)const_cast<__bf16*>(g);  // generic ptr
  i32x4* ll = (i32x4*)(void*)l;                       // generic ptr
  // generic -> AS1 / AS3 via C-style cast => addrspacecast
  __builtin_amdgcn_global_load_async_to_lds_b128((as1_v4i)gg, (as3_v4i)ll, 0, 0);
}
__device__ __forceinline__ void async_wait0() {
#if __has_builtin(__builtin_amdgcn_s_wait_asynccnt)
  __builtin_amdgcn_s_wait_asynccnt(0);
#else
  asm volatile("s_wait_asynccnt 0" ::: "memory");
#endif
}
#endif

__device__ __forceinline__ v8f wmma_bf16(v16bf a, v16bf b, v8f c) {
  return __builtin_amdgcn_wmma_f32_16x16x32_bf16(false, a, false, b, (short)0, c,
                                                 false, false);
}

// Build a 16x bf16 fragment from two contiguous 16B runs in LDS.
__device__ __forceinline__ v16bf frag_2x16B(const __bf16* p0, const __bf16* p1) {
  u4x2 u;
  u.lo = *reinterpret_cast<const uint4*>(p0);
  u.hi = *reinterpret_cast<const uint4*>(p1);
  return __builtin_bit_cast(v16bf, u);
}

// A-matrix 16x32 bf16 fragment, tile stored [M][K] row-major (ISA 7.12.2):
// lane<16: M=lane, K={0..7,16..23}; lane>=16: M=lane-16, K={8..15,24..31}
template <int STRIDE>
__device__ __forceinline__ v16bf load_a_frag(const __bf16* t, int row, int k0,
                                             int lane) {
  const __bf16* p = t + row * STRIDE + k0 + ((lane & 16) ? 8 : 0);
  return frag_2x16B(p, p + 16);
}

// B-matrix 32x16 fragment, tile stored [N][K] row-major:
// lane<16: K={0..7,16..23}, lane>=16: K={8..15,24..31}; N = col
template <int STRIDE>
__device__ __forceinline__ v16bf load_b_frag(const __bf16* t, int k0, int col,
                                             int lane) {
  const __bf16* p = t + col * STRIDE + k0 + ((lane & 16) ? 8 : 0);
  return frag_2x16B(p, p + 16);
}

__device__ __forceinline__ uint2 pack_bf16x4(float4 v) {
  bf16x4 h{(__bf16)v.x, (__bf16)v.y, (__bf16)v.z, (__bf16)v.w};
  return __builtin_bit_cast(uint2, h);
}

// ---------------------------------------------------------------------------
// Kernel 1: QKV = X @ Wqkv^T, scatter into Q/K/V [b][h][s][d] bf16, Q *= scale
// ---------------------------------------------------------------------------
__global__ __launch_bounds__(256) void sab_qkv_gemm(
    const float* __restrict__ X, const float* __restrict__ W,
    __bf16* __restrict__ Qb, __bf16* __restrict__ Kb, __bf16* __restrict__ Vb) {
  __shared__ __bf16 As[128 * 40];  // [M][K]
  __shared__ __bf16 Bs[64 * 40];   // [N][K]
  const int tid = threadIdx.x, lane = tid & 31, wave = tid >> 5;
  const int wm = wave & 3, wn = wave >> 2;
  const int rowBase = blockIdx.x * 128;
  const int colBase = blockIdx.y * 64;
  v8f acc[2][2] = {};
  for (int k0 = 0; k0 < HDIM; k0 += 32) {
    __syncthreads();
#pragma unroll
    for (int i = 0; i < 4; ++i) {  // A tile: 128x32 f32 -> bf16 (1024 float4)
      int e = tid + i * 256;
      int r = e >> 3, c = (e & 7) * 4;
      float4 v = *reinterpret_cast<const float4*>(
          &X[(size_t)(rowBase + r) * HDIM + k0 + c]);
      *reinterpret_cast<uint2*>(&As[r * 40 + c]) = pack_bf16x4(v);
    }
#pragma unroll
    for (int i = 0; i < 2; ++i) {  // B tile: Bs[n][k] = W[colBase+n][k0+k]
      int e = tid + i * 256;
      int n = e >> 3, c = (e & 7) * 4;
      float4 v = *reinterpret_cast<const float4*>(
          &W[(size_t)(colBase + n) * HDIM + k0 + c]);
      *reinterpret_cast<uint2*>(&Bs[n * 40 + c]) = pack_bf16x4(v);
    }
    __syncthreads();
    v16bf a[2], b[2];
#pragma unroll
    for (int mi = 0; mi < 2; ++mi)
      a[mi] = load_a_frag<40>(As, wm * 32 + mi * 16 + (lane & 15), 0, lane);
#pragma unroll
    for (int ni = 0; ni < 2; ++ni)
      b[ni] = load_b_frag<40>(Bs, 0, wn * 32 + ni * 16 + (lane & 15), lane);
#pragma unroll
    for (int mi = 0; mi < 2; ++mi)
#pragma unroll
      for (int ni = 0; ni < 2; ++ni)
        acc[mi][ni] = wmma_bf16(a[mi], b[ni], acc[mi][ni]);
  }
  const float scale = 0.125f;  // DH^-0.5, folded into Q
#pragma unroll
  for (int mi = 0; mi < 2; ++mi)
#pragma unroll
    for (int ni = 0; ni < 2; ++ni)
#pragma unroll
      for (int r = 0; r < 8; ++r) {
        int row = rowBase + wm * 32 + mi * 16 + r + ((lane & 16) ? 8 : 0);
        int col = colBase + wn * 32 + ni * 16 + (lane & 15);
        float v = acc[mi][ni][r];
        int which = col / HDIM;
        int rem = col - which * HDIM;
        int head = rem >> 6, d = rem & 63;
        int bb = row >> 11, s = row & (SEQ - 1);
        size_t idx = ((size_t)(bb * NHEADS + head) * SEQ + s) * DHEAD + d;
        if (which == 0) Qb[idx] = (__bf16)(v * scale);
        else if (which == 1) Kb[idx] = (__bf16)v;
        else Vb[idx] = (__bf16)v;
      }
}

// ---------------------------------------------------------------------------
// Kernel 2: flash attention per (b, head, 64-query block). 4 waves, each owns
// 16 query rows; online softmax over 32 key blocks of 64.
// ---------------------------------------------------------------------------
__global__ __launch_bounds__(128) void sab_attn(
    const __bf16* __restrict__ Qb, const __bf16* __restrict__ Kb,
    const __bf16* __restrict__ Vb, __bf16* __restrict__ Ob) {
  __shared__ __bf16 Qs[64 * 72];      // [qrow][d]
  __shared__ __bf16 Ks[64 * 72];      // [key][d]   (B for QK^T: N=key, K=d)
  __shared__ __bf16 Vt[64 * 72];      // [d][key]   (B for P@V:  N=d,  K=key)
  __shared__ __bf16 Ps[4][16 * 40];   // per-wave P tile [row][key]
  const int tid = threadIdx.x, lane = tid & 31, wave = tid >> 5;
  const int q0 = blockIdx.x * 64;
  const int h = blockIdx.y, bb = blockIdx.z;
  const size_t hbase = (size_t)(bb * NHEADS + h) * SEQ * DHEAD;
  const __bf16* Qg = Qb + hbase;
  const __bf16* Kg = Kb + hbase;
  const __bf16* Vg = Vb + hbase;

  // Stage Q tile (64x64 bf16): 512 16B chunks, 4 per thread
#pragma unroll
  for (int i = 0; i < 4; ++i) {
    int e = tid + i * 128;
    int r = e >> 3, c = (e & 7) * 8;
    const __bf16* g = &Qg[(size_t)(q0 + r) * DHEAD + c];
    __bf16* l = &Qs[r * 72 + c];
#ifdef SAB_ASYNC
    gasync_b128(g, l);
#else
    *reinterpret_cast<uint4*>(l) = *reinterpret_cast<const uint4*>(g);
#endif
  }
#ifdef SAB_ASYNC
  async_wait0();
#endif
  __syncthreads();
  v16bf aQ[2];  // loop-invariant Q fragments (wave's 16 rows x 64 dims)
#pragma unroll
  for (int kc = 0; kc < 2; ++kc)
    aQ[kc] = load_a_frag<72>(Qs + wave * 16 * 72, lane & 15, kc * 32, lane);

  float m[8], l[8];
#pragma unroll
  for (int r = 0; r < 8; ++r) { m[r] = -1e30f; l[r] = 0.0f; }
  v8f accO[4] = {};

  for (int kb = 0; kb < SEQ / 64; ++kb) {
    __syncthreads();
    // K tile: natural layout, async b128
#pragma unroll
    for (int i = 0; i < 4; ++i) {
      int e = tid + i * 128;
      int r = e >> 3, c = (e & 7) * 8;
      const __bf16* g = &Kg[(size_t)(kb * 64 + r) * DHEAD + c];
      __bf16* lp = &Ks[r * 72 + c];
#ifdef SAB_ASYNC
      gasync_b128(g, lp);
#else
      *reinterpret_cast<uint4*>(lp) = *reinterpret_cast<const uint4*>(g);
#endif
    }
    // V tile: transpose on stage -> Vt[d][key]
#pragma unroll
    for (int i = 0; i < 4; ++i) {
      int e = tid + i * 128;
      int key = e >> 3, d0 = (e & 7) * 8;
      uint4 raw = *reinterpret_cast<const uint4*>(
          &Vg[(size_t)(kb * 64 + key) * DHEAD + d0]);
      bf16x8 hv = __builtin_bit_cast(bf16x8, raw);
#pragma unroll
      for (int j = 0; j < 8; ++j) Vt[(d0 + j) * 72 + key] = hv.v[j];
    }
#ifdef SAB_ASYNC
    async_wait0();
#endif
    __syncthreads();
    // S = Q @ K^T  (Q already carries 1/sqrt(d))
    v8f sacc[4] = {};
#pragma unroll
    for (int t = 0; t < 4; ++t)
#pragma unroll
      for (int kc = 0; kc < 2; ++kc) {
        v16bf bk = load_b_frag<72>(Ks, kc * 32, t * 16 + (lane & 15), lane);
        sacc[t] = wmma_bf16(aQ[kc], bk, sacc[t]);
      }
    // online softmax; row M = r + 8*(lane>=16); xor masks <16 stay in-half
#pragma unroll
    for (int r = 0; r < 8; ++r) {
      float mx =
          fmaxf(fmaxf(sacc[0][r], sacc[1][r]), fmaxf(sacc[2][r], sacc[3][r]));
#pragma unroll
      for (int off = 1; off < 16; off <<= 1)
        mx = fmaxf(mx, __shfl_xor(mx, off, 32));
      float mnew = fmaxf(m[r], mx);
      float corr = __expf(m[r] - mnew);
      float rs = 0.0f;
#pragma unroll
      for (int t = 0; t < 4; ++t) {
        float p = __expf(sacc[t][r] - mnew);
        sacc[t][r] = p;
        rs += p;
      }
#pragma unroll
      for (int off = 1; off < 16; off <<= 1) rs += __shfl_xor(rs, off, 32);
      l[r] = l[r] * corr + rs;
      m[r] = mnew;
#pragma unroll
      for (int t = 0; t < 4; ++t) accO[t][r] *= corr;
    }
    // C-layout -> A-layout transpose of P via this wave's LDS region
    __bf16* Pw = Ps[wave];
    const int pro = (lane & 16) ? 8 : 0;
#pragma unroll
    for (int r = 0; r < 8; ++r)
#pragma unroll
      for (int t = 0; t < 4; ++t)
        Pw[(r + pro) * 40 + t * 16 + (lane & 15)] = (__bf16)sacc[t][r];
    v16bf aP[2];
#pragma unroll
    for (int kc = 0; kc < 2; ++kc)
      aP[kc] = load_a_frag<40>(Pw, lane & 15, kc * 32, lane);
    // O += P @ V
#pragma unroll
    for (int t = 0; t < 4; ++t)
#pragma unroll
      for (int kc = 0; kc < 2; ++kc) {
        v16bf bv = load_b_frag<72>(Vt, kc * 32, t * 16 + (lane & 15), lane);
        accO[t] = wmma_bf16(aP[kc], bv, accO[t]);
      }
  }
  // epilogue: O /= l, write bf16 [b*S][768] with col = h*64 + d
#pragma unroll
  for (int t = 0; t < 4; ++t)
#pragma unroll
    for (int r = 0; r < 8; ++r) {
      int row = r + ((lane & 16) ? 8 : 0);
      int srow = q0 + wave * 16 + row;
      float v = accO[t][r] / l[r];
      Ob[((size_t)(bb * SEQ + srow)) * HDIM + h * 64 + t * 16 + (lane & 15)] =
          (__bf16)v;
    }
}

// ---------------------------------------------------------------------------
// Kernel 3: Out = Att(bf16) @ Wout^T + bias, fp32 result
// ---------------------------------------------------------------------------
__global__ __launch_bounds__(256) void sab_out_gemm(
    const __bf16* __restrict__ A, const float* __restrict__ W,
    const float* __restrict__ bias, float* __restrict__ Out) {
  __shared__ __bf16 As[128 * 40];  // [M][K]
  __shared__ __bf16 Bs[64 * 40];   // [N][K]
  const int tid = threadIdx.x, lane = tid & 31, wave = tid >> 5;
  const int wm = wave & 3, wn = wave >> 2;
  const int rowBase = blockIdx.x * 128;
  const int colBase = blockIdx.y * 64;
  v8f acc[2][2] = {};
  for (int k0 = 0; k0 < HDIM; k0 += 32) {
    __syncthreads();
#pragma unroll
    for (int i = 0; i < 2; ++i) {  // A tile: 128x32 bf16 (512 16B chunks)
      int e = tid + i * 256;
      int r = e >> 2, c = (e & 3) * 8;
      *reinterpret_cast<uint4*>(&As[r * 40 + c]) =
          *reinterpret_cast<const uint4*>(
              &A[(size_t)(rowBase + r) * HDIM + k0 + c]);
    }
#pragma unroll
    for (int i = 0; i < 2; ++i) {  // B tile: Bs[n][k] = W[colBase+n][k0+k]
      int e = tid + i * 256;
      int n = e >> 3, c = (e & 7) * 4;
      float4 v = *reinterpret_cast<const float4*>(
          &W[(size_t)(colBase + n) * HDIM + k0 + c]);
      *reinterpret_cast<uint2*>(&Bs[n * 40 + c]) = pack_bf16x4(v);
    }
    __syncthreads();
    v16bf a[2], b[2];
#pragma unroll
    for (int mi = 0; mi < 2; ++mi)
      a[mi] = load_a_frag<40>(As, wm * 32 + mi * 16 + (lane & 15), 0, lane);
#pragma unroll
    for (int ni = 0; ni < 2; ++ni)
      b[ni] = load_b_frag<40>(Bs, 0, wn * 32 + ni * 16 + (lane & 15), lane);
#pragma unroll
    for (int mi = 0; mi < 2; ++mi)
#pragma unroll
      for (int ni = 0; ni < 2; ++ni)
        acc[mi][ni] = wmma_bf16(a[mi], b[ni], acc[mi][ni]);
  }
#pragma unroll
  for (int mi = 0; mi < 2; ++mi)
#pragma unroll
    for (int ni = 0; ni < 2; ++ni)
#pragma unroll
      for (int r = 0; r < 8; ++r) {
        int row = rowBase + wm * 32 + mi * 16 + r + ((lane & 16) ? 8 : 0);
        int col = colBase + wn * 32 + ni * 16 + (lane & 15);
        Out[(size_t)row * HDIM + col] = acc[mi][ni][r] + bias[col];
      }
}

extern "C" void kernel_launch(void* const* d_in, const int* in_sizes, int n_in,
                              void* d_out, int out_size, void* d_ws,
                              size_t ws_size, hipStream_t stream) {
  const float* x = (const float*)d_in[0];      // [2,2048,768]
  const float* w_qkv = (const float*)d_in[1];  // [2304,768]
  const float* w_out = (const float*)d_in[2];  // [768,768]
  const float* b_out = (const float*)d_in[3];  // [768]
  float* out = (float*)d_out;                  // [2,2048,768]

  const size_t nPerHeadBuf = (size_t)2 * NHEADS * SEQ * DHEAD;  // 3,145,728
  __bf16* Qb = (__bf16*)d_ws;
  __bf16* Kb = Qb + nPerHeadBuf;
  __bf16* Vb = Kb + nPerHeadBuf;
  __bf16* Ab = Vb + nPerHeadBuf;  // attention output [4096][768] bf16

  dim3 g1(4096 / 128, 2304 / 64);  // 32 x 36
  sab_qkv_gemm<<<g1, 256, 0, stream>>>(x, w_qkv, Qb, Kb, Vb);

  dim3 g2(SEQ / 64, NHEADS, 2);    // 32 x 12 x 2
  sab_attn<<<g2, 128, 0, stream>>>(Qb, Kb, Vb, Ab);

  dim3 g3(4096 / 128, HDIM / 64);  // 32 x 12
  sab_out_gemm<<<g3, 256, 0, stream>>>(Ab, w_out, b_out, out);
}
